// MambaEncoder_3702261809514
// MI455X (gfx1250) — compile-verified, matrix operand order checked by reading, not run on hardware
//
#include <hip/hip_runtime.h>
#include <hip/hip_bf16.h>
#include <math.h>

// ---------------------------------------------------------------------------
// MambaEncoder for MI455X (gfx1250): all GEMMs via v_wmma_f32_16x16x32_bf16,
// selective scan as register-resident recurrence with ASYNC global->LDS
// staging (global_load_async_to_lds_b64 + s_wait_asynccnt), fused epilogues.
// ---------------------------------------------------------------------------

typedef __attribute__((ext_vector_type(16))) __bf16 v16bf;
typedef __attribute__((ext_vector_type(8)))  float  v8f;

#define NLYR 8
#define DMOD 512
#define DINN 1024
#define DSN  16
#define DCV  4
#define DRK  32
#define BBAT 8
#define LSEQ 800
#define MROWS (BBAT*LSEQ)   // 6400

__device__ __forceinline__ unsigned short f2bf(float f) {
  __bf16 h = (__bf16)f;                       // native v_cvt to bf16
  return __builtin_bit_cast(unsigned short, h);
}

union Frag16 { v16bf v; unsigned u[8]; };

// ----------------------------- WMMA GEMM -----------------------------------
// C[M x N] = act(A[M x K] * W[K x N] + bias). M=6400 (mult of 128),
// N mult of 64, K mult of 32. ACT: 0=none, 1=softplus, 2=gelu(exact).
#define BM 128
#define BN 64
#define BK 32
#define LSTR (BK + 2)

template<int ACT, bool HAS_BIAS>
__global__ __launch_bounds__(256)
void gemm_bf16_wmma(const float* __restrict__ A, int lda,
                    const float* __restrict__ W, int ldw,
                    float* __restrict__ C, int ldc,
                    const float* __restrict__ bias, int K)
{
  __shared__ unsigned short lsA[BM][LSTR];   // A tile, row-major  [m][k]
  __shared__ unsigned short lsB[BN][LSTR];   // W tile, transposed [n][k]
  const int tid  = threadIdx.x;
  const int m0   = blockIdx.y * BM;
  const int n0   = blockIdx.x * BN;
  const int wid  = tid >> 5;
  const int lane = tid & 31;
  const int wm   = (wid & 3) * 32;           // 4 waves along M
  const int wn   = (wid >> 2) * 32;          // 2 waves along N
  const int lrow  = lane & 15;
  const int lhalf = lane >> 4;

  v8f acc00 = {}, acc01 = {}, acc10 = {}, acc11 = {};

  for (int k0 = 0; k0 < K; k0 += BK) {
    if (k0 + BK < K)                          // uniform: keeps EXEC all-ones
      __builtin_prefetch(&A[(long)(m0 + (tid >> 1))*lda + k0 + BK], 0, 0);
    #pragma unroll
    for (int i = 0; i < (BM*BK)/256; ++i) {   // stage A (f32 -> bf16)
      int e = tid + i*256;
      int m = e >> 5, k = e & 31;
      lsA[m][k] = f2bf(A[(long)(m0+m)*lda + k0 + k]);
    }
    #pragma unroll
    for (int i = 0; i < (BN*BK)/256; ++i) {   // stage W transposed
      int e = tid + i*256;
      int k = e >> 6, n = e & 63;
      lsB[n][k] = f2bf(W[(long)(k0+k)*ldw + n0 + n]);
    }
    __syncthreads();

    // Fragment fill per ISA 7.12.2 (16-bit A 16x32); B as column-major dual.
    Frag16 a0, a1, b0, b1;
    #pragma unroll
    for (int v = 0; v < 8; ++v) {
      int kb = (v < 4) ? (lhalf*8 + 2*v) : (16 + lhalf*8 + 2*(v-4));
      a0.u[v] = *(const unsigned*)&lsA[wm + lrow][kb];
      a1.u[v] = *(const unsigned*)&lsA[wm + 16 + lrow][kb];
      b0.u[v] = *(const unsigned*)&lsB[wn + lrow][kb];
      b1.u[v] = *(const unsigned*)&lsB[wn + 16 + lrow][kb];
    }
    acc00 = __builtin_amdgcn_wmma_f32_16x16x32_bf16(false, a0.v, false, b0.v, (short)0, acc00, false, false);
    acc01 = __builtin_amdgcn_wmma_f32_16x16x32_bf16(false, a0.v, false, b1.v, (short)0, acc01, false, false);
    acc10 = __builtin_amdgcn_wmma_f32_16x16x32_bf16(false, a1.v, false, b0.v, (short)0, acc10, false, false);
    acc11 = __builtin_amdgcn_wmma_f32_16x16x32_bf16(false, a1.v, false, b1.v, (short)0, acc11, false, false);
    __syncthreads();
  }

  #pragma unroll
  for (int sm = 0; sm < 2; ++sm) {
    #pragma unroll
    for (int sn = 0; sn < 2; ++sn) {
      v8f c = (sm==0) ? (sn==0 ? acc00 : acc01) : (sn==0 ? acc10 : acc11);
      int n = n0 + wn + sn*16 + lrow;
      float bv = HAS_BIAS ? bias[n] : 0.0f;
      #pragma unroll
      for (int r = 0; r < 8; ++r) {
        int m = m0 + wm + sm*16 + lhalf*8 + r;   // C layout: M = r + 8*half
        float x = c[r] + bv;
        if (ACT == 1)      x = (x > 20.0f) ? x : log1pf(__expf(x));        // softplus
        else if (ACT == 2) x = 0.5f * x * (1.0f + erff(x * 0.70710678f));  // gelu exact
        C[(long)m*ldc + n] = x;
      }
    }
  }
}

// ------------------------- patch rearrange ---------------------------------
// x (B,1,32,160,160) -> xp (B, 800, 1024), feature = (p1,p2,pf)
__global__ void patch_rearrange(const float* __restrict__ x, float* __restrict__ xp)
{
  int idx = blockIdx.x*256 + threadIdx.x;
  if (idx >= BBAT*LSEQ*1024) return;
  int fidx = idx & 1023;
  int tok  = (idx >> 10) % LSEQ;
  int b    = idx / (LSEQ*1024);
  int f = tok / 100, rem = tok % 100, h = rem / 10, w = rem % 10;
  int p1 = fidx >> 6, r2 = fidx & 63, p2 = r2 >> 2, pf = r2 & 3;
  int ft = f*4 + pf, row = h*16 + p1, col = w*16 + p2;
  xp[idx] = x[((long)(b*32 + ft)*160 + row)*160 + col];
}

// ------------------- depthwise causal conv (K=4) + SiLU --------------------
// xz row stride 2*DINN (xm = first DINN cols). rev=1: operate on reversed seq.
__global__ void conv_silu(const float* __restrict__ xz, const float* __restrict__ w,
                          const float* __restrict__ bias, float* __restrict__ out, int rev)
{
  int idx = blockIdx.x*256 + threadIdx.x;
  if (idx >= BBAT*LSEQ*DINN) return;
  int d = idx & (DINN-1);
  int t = (idx >> 10) % LSEQ;
  int b = idx / (DINN*LSEQ);
  float acc = bias[d];
  #pragma unroll
  for (int j = 0; j < DCV; ++j) {
    int ts = t - (DCV-1) + j;
    if (ts >= 0) {
      int tsrc = rev ? (LSEQ-1-ts) : ts;
      acc += w[d*DCV + j] * xz[((long)(b*LSEQ + tsrc))*(2*DINN) + d];
    }
  }
  out[idx] = acc / (1.0f + __expf(-acc));      // SiLU
}

// --------------------------- selective scan --------------------------------
// One lane per (b,d): h[16] and a[16]=-exp(A_log) live in VGPRs. The shared
// per-step [B(16)|C(16)] vectors (128 B/row) are staged to LDS with the
// gfx1250 ASYNC copy path: each of the 8 waves issues one
// global_load_async_to_lds_b64 (32 lanes x 8 B = rows 2w,2w+1), waits its
// ASYNCcnt, then the workgroup barrier publishes all 16 rows.
#define TCH 16
__global__ __launch_bounds__(256)
void ssm_scan(const float* __restrict__ u_in, float* __restrict__ y_out,
              const float* __restrict__ dt, const float* __restrict__ proj,
              const float* __restrict__ A_log, const float* __restrict__ Dv)
{
  __shared__ float sBC[TCH][2*DSN];            // [tt][ B(16) | C(16) ]
  const int tid  = threadIdx.x;
  const int wid  = tid >> 5;
  const int lane = tid & 31;
  int b = blockIdx.x >> 2;
  int d = ((blockIdx.x & 3) << 8) + tid;

  float a[DSN], h[DSN];
  #pragma unroll
  for (int n = 0; n < DSN; ++n) { a[n] = -__expf(A_log[d*DSN + n]); h[n] = 0.0f; }
  float Dval = Dv[d];
  const long rowbase = (long)b * LSEQ;

  // async-copy addressing (fixed per thread): this wave covers rows 2w,2w+1
  const int myrow = 2*wid + (lane >> 4);             // 0..15
  const int mybyte = (lane & 15) * 8;                // 0..120
  const unsigned lds0 = (unsigned)(unsigned long long)&sBC[0][0];  // low 32 bits are the LDS address
  const unsigned ldsaddr = lds0 + (unsigned)(myrow * (2*DSN*4) + mybyte);
  const unsigned long long gbase =
      (unsigned long long)(const void*)(proj + rowbase*64 + DRK);  // B|C start of row 0

  for (int t0 = 0; t0 < LSEQ; t0 += TCH) {
    unsigned voff = (unsigned)((t0 + myrow) * 64 * 4 + mybyte);
    asm volatile("global_load_async_to_lds_b64 %0, %1, %2"
                 :: "v"(ldsaddr), "v"(voff), "s"(gbase) : "memory");
    asm volatile("s_wait_asynccnt 0" ::: "memory");
    __syncthreads();
    #pragma unroll 4
    for (int tt = 0; tt < TCH; ++tt) {
      long off = (rowbase + t0 + tt)*DINN + d;
      float dtv = dt[off];
      float uv  = u_in[off];
      float du  = dtv * uv;
      float acc = 0.0f;
      #pragma unroll
      for (int n = 0; n < DSN; ++n) {
        h[n] = __expf(dtv * a[n]) * h[n] + du * sBC[tt][n];
        acc += h[n] * sBC[tt][DSN + n];
      }
      y_out[off] = acc + uv * Dval;
    }
    __syncthreads();
  }
}

// -------------------- combine fwd/bwd + SiLU(z) gate -----------------------
// yb is in reversed-time order; yb[::-1] re-aligns gate to silu(z[t]) for
// both directions. Result written into xm slot of xz (lda=2048 for out_proj).
__global__ void combine_gate(const float* __restrict__ yf, const float* __restrict__ yb,
                             float* __restrict__ xz)
{
  int idx = blockIdx.x*256 + threadIdx.x;
  if (idx >= BBAT*LSEQ*DINN) return;
  int d = idx & (DINN-1);
  int t = (idx >> 10) % LSEQ;
  int b = idx / (DINN*LSEQ);
  long row = (long)b*LSEQ + t;
  float z  = xz[row*(2*DINN) + DINN + d];
  float sz = z / (1.0f + __expf(-z));
  float v  = (yf[row*DINN + d] + yb[((long)b*LSEQ + (LSEQ-1-t))*DINN + d]) * sz;
  xz[row*(2*DINN) + d] = v;
}

// ---------------------- LayerNorm + residual (wave/row) --------------------
__global__ __launch_bounds__(256)
void ln_residual(const float* __restrict__ y, const float* __restrict__ g,
                 const float* __restrict__ beta, float* __restrict__ xt,
                 float* __restrict__ out2)
{
  int wid = threadIdx.x >> 5, lane = threadIdx.x & 31;
  long row = (long)blockIdx.x * 8 + wid;
  const float* yr = y + row * DMOD;
  float vals[16];
  float s = 0.0f;
  #pragma unroll
  for (int i = 0; i < 16; ++i) { vals[i] = yr[lane + 32*i]; s += vals[i]; }
  #pragma unroll
  for (int off = 16; off >= 1; off >>= 1) s += __shfl_xor(s, off, 32);
  float mean = s * (1.0f/DMOD);
  float vs = 0.0f;
  #pragma unroll
  for (int i = 0; i < 16; ++i) { float dd = vals[i]-mean; vs += dd*dd; }
  #pragma unroll
  for (int off = 16; off >= 1; off >>= 1) vs += __shfl_xor(vs, off, 32);
  float rstd = rsqrtf(vs * (1.0f/DMOD) + 1e-5f);
  #pragma unroll
  for (int i = 0; i < 16; ++i) {
    int c = lane + 32*i;
    float o = (vals[i]-mean)*rstd*g[c] + beta[c] + xt[row*DMOD + c];
    xt[row*DMOD + c] = o;
    if (out2) out2[row*DMOD + c] = o;
  }
}

// ---------------------------------------------------------------------------
extern "C" void kernel_launch(void* const* d_in, const int* in_sizes, int n_in,
                              void* d_out, int out_size, void* d_ws, size_t ws_size,
                              hipStream_t stream)
{
  (void)in_sizes; (void)n_in; (void)out_size; (void)ws_size;
  const float* x         = (const float*)d_in[0];
  const float* patch_w   = (const float*)d_in[1];
  const float* patch_b   = (const float*)d_in[2];
  const float* in_proj_w = (const float*)d_in[3];
  const float* conv_w    = (const float*)d_in[4];
  const float* conv_b    = (const float*)d_in[5];
  const float* xproj_w   = (const float*)d_in[6];
  const float* dt_w      = (const float*)d_in[7];
  const float* dt_bias   = (const float*)d_in[8];
  const float* A_log     = (const float*)d_in[9];
  const float* D_skip    = (const float*)d_in[10];
  const float* conv_w_b  = (const float*)d_in[11];
  const float* conv_b_b  = (const float*)d_in[12];
  const float* xproj_w_b = (const float*)d_in[13];
  const float* dt_w_b    = (const float*)d_in[14];
  const float* dt_bias_b = (const float*)d_in[15];
  const float* A_log_b   = (const float*)d_in[16];
  const float* D_skip_b  = (const float*)d_in[17];
  const float* out_proj_w= (const float*)d_in[18];
  const float* ln1_g     = (const float*)d_in[19];
  const float* ln1_b     = (const float*)d_in[20];
  const float* fc1_w     = (const float*)d_in[21];
  const float* fc1_b     = (const float*)d_in[22];
  const float* fc2_w     = (const float*)d_in[23];
  const float* fc2_b     = (const float*)d_in[24];
  const float* ln2_g     = (const float*)d_in[25];
  const float* ln2_b     = (const float*)d_in[26];
  float* out = (float*)d_out;

  // workspace layout (floats) ~152 MB, fully L2-resident per-buffer
  float* ws = (float*)d_ws;
  const long SZ_XT = (long)MROWS*DMOD;        // 3,276,800
  const long SZ_XZ = (long)MROWS*2*DINN;      // 13,107,200
  const long SZ_DI = (long)MROWS*DINN;        // 6,553,600
  const long SZ_PJ = (long)MROWS*64;          // 409,600
  float* xt   = ws;
  float* xz   = xt  + SZ_XT;
  float* xcf  = xz  + SZ_XZ;    // fwd conv out -> fwd scan out (in place)
  float* xcb  = xcf + SZ_DI;    // bwd (reversed) conv out -> bwd scan out
  float* dtb  = xcb + SZ_DI;    // dt (fwd then bwd), later MLP hidden
  float* proj = dtb + SZ_DI;    // x_proj output (fwd then bwd)
  float* yout = proj + SZ_PJ;   // out_proj / fc2 result
  float* xpatch = xcf;          // alias: only live before the layer loop

  dim3 blk(256);
  const int gE = (MROWS*DINN + 255)/256;      // elementwise grids

  // patch embed: rearrange then GEMM(+bias) -> xt (6400 x 512)
  patch_rearrange<<<gE, blk, 0, stream>>>(x, xpatch);
  gemm_bf16_wmma<0,true><<<dim3(DMOD/BN, MROWS/BM), blk, 0, stream>>>(
      xpatch, DINN, patch_w, DMOD, xt, DMOD, patch_b, DINN);

  for (int l = 0; l < NLYR; ++l) {
    // in_proj: xt @ (512x2048) -> xz (xm | z)
    gemm_bf16_wmma<0,false><<<dim3(2*DINN/BN, MROWS/BM), blk, 0, stream>>>(
        xt, DMOD, in_proj_w + (long)l*DMOD*2*DINN, 2*DINN, xz, 2*DINN, nullptr, DMOD);
    // depthwise causal conv + SiLU, both directions
    conv_silu<<<gE, blk, 0, stream>>>(xz, conv_w   + l*DINN*DCV, conv_b   + l*DINN, xcf, 0);
    conv_silu<<<gE, blk, 0, stream>>>(xz, conv_w_b + l*DINN*DCV, conv_b_b + l*DINN, xcb, 1);
    // forward: x_proj -> dt (softplus fused) -> scan (in place on xcf)
    gemm_bf16_wmma<0,false><<<dim3(64/BN, MROWS/BM), blk, 0, stream>>>(
        xcf, DINN, xproj_w + (long)l*DINN*64, 64, proj, 64, nullptr, DINN);
    gemm_bf16_wmma<1,true><<<dim3(DINN/BN, MROWS/BM), blk, 0, stream>>>(
        proj, 64, dt_w + (long)l*DRK*DINN, DINN, dtb, DINN, dt_bias + l*DINN, DRK);
    ssm_scan<<<32, blk, 0, stream>>>(xcf, xcf, dtb, proj,
                                     A_log + (long)l*DINN*DSN, D_skip + l*DINN);
    // backward (reuses proj/dtb buffers)
    gemm_bf16_wmma<0,false><<<dim3(64/BN, MROWS/BM), blk, 0, stream>>>(
        xcb, DINN, xproj_w_b + (long)l*DINN*64, 64, proj, 64, nullptr, DINN);
    gemm_bf16_wmma<1,true><<<dim3(DINN/BN, MROWS/BM), blk, 0, stream>>>(
        proj, 64, dt_w_b + (long)l*DRK*DINN, DINN, dtb, DINN, dt_bias_b + l*DINN, DRK);
    ssm_scan<<<32, blk, 0, stream>>>(xcb, xcb, dtb, proj,
                                     A_log_b + (long)l*DINN*DSN, D_skip_b + l*DINN);
    // combine + SiLU(z) gate -> xm slot of xz
    combine_gate<<<gE, blk, 0, stream>>>(xcf, xcb, xz);
    // out_proj (A = xz with lda=2048, first 1024 cols)
    gemm_bf16_wmma<0,false><<<dim3(DMOD/BN, MROWS/BM), blk, 0, stream>>>(
        xz, 2*DINN, out_proj_w + (long)l*DINN*DMOD, DMOD, yout, DMOD, nullptr, DINN);
    // ln1 + residual
    ln_residual<<<MROWS/8, blk, 0, stream>>>(yout, ln1_g + l*DMOD, ln1_b + l*DMOD, xt, nullptr);
    // MLP: fc1(+bias, gelu) -> hidden (reuse dtb), fc2(+bias) -> yout
    gemm_bf16_wmma<2,true><<<dim3(1024/BN, MROWS/BM), blk, 0, stream>>>(
        xt, DMOD, fc1_w + (long)l*DMOD*1024, 1024, dtb, 1024, fc1_b + l*1024, DMOD);
    gemm_bf16_wmma<0,true><<<dim3(DMOD/BN, MROWS/BM), blk, 0, stream>>>(
        dtb, 1024, fc2_w + (long)l*1024*DMOD, DMOD, yout, DMOD, fc2_b + l*DMOD, 1024);
    // ln2 + residual, and emit this layer's output slice
    ln_residual<<<MROWS/8, blk, 0, stream>>>(yout, ln2_g + l*DMOD, ln2_b + l*DMOD,
                                             xt, out + (long)l*SZ_XT);
  }
}
